// PyGRandLANet_53283364274314
// MI455X (gfx1250) — matrix operand order repaired and checked by reading, not compiled, not run on hardware
//
#include <hip/hip_runtime.h>
#include <hip/hip_bf16.h>

// ---------------------------------------------------------------------------
// MI455X (gfx1250) RandLA-Net forward.
// All matmuls -> v_wmma_f32_16x16x32_f16 (f16 in, f32 accum).
// wave32: blockDim 256 = 8 waves; GEMM workgroup tile 128(M) x 64(N),
// wave grid 4x2, each wave owns a 32x32 output block = 4 WMMA per K-step.
// Interior tiles staged to LDS with global_load_async_to_lds_b128 (ASYNCcnt)
// when the builtin is available, else vectorized b128 copies.
// ---------------------------------------------------------------------------

typedef __attribute__((ext_vector_type(16))) _Float16 v16h;
typedef __attribute__((ext_vector_type(8)))  float    v8f;
typedef __attribute__((ext_vector_type(4)))  int      v4i;

#define NEG_SLOPE 0.2f
#define BN_EPS    1e-6f
#define KNN_K     16

#if defined(__gfx1250__) && __has_builtin(__builtin_amdgcn_global_load_async_to_lds_b128) && \
    __has_builtin(__builtin_amdgcn_s_wait_asynccnt)
#define USE_ASYNC_LDS 1
#define ASYNC_COPY_B128(dst_lds, src_glb)                                      \
  __builtin_amdgcn_global_load_async_to_lds_b128(                              \
      (__attribute__((address_space(1))) v4i*)(src_glb),                       \
      (__attribute__((address_space(3))) v4i*)(dst_lds), 0, 0)
#else
#define USE_ASYNC_LDS 0
#endif

static __device__ __forceinline__ float leaky(float v) {
  return v > 0.f ? v : NEG_SLOPE * v;
}

// ---------------------------------------------------------------------------
// dtype conversion helpers
// ---------------------------------------------------------------------------
__global__ void f32_to_f16_k(const float* __restrict__ s, _Float16* __restrict__ d, long long n) {
  long long i = (long long)blockIdx.x * blockDim.x + threadIdx.x;
  if (i < n) d[i] = (_Float16)s[i];
}

// W [K,N] row-major f32 -> Wt [N,K] f16 (column-of-W contiguous in K)
__global__ void w_to_f16t_k(const float* __restrict__ W, _Float16* __restrict__ Wt, int K, int N) {
  long long i = (long long)blockIdx.x * blockDim.x + threadIdx.x;
  if (i >= (long long)K * N) return;
  int k = (int)(i / N), n = (int)(i % N);
  Wt[(long long)n * K + k] = (_Float16)W[i];
}

// ---------------------------------------------------------------------------
// WMMA GEMM: C[M,N] = A[M,K] * B[K,N] (+bias), A f16 row-major, Bt f16 = B^T
// ---------------------------------------------------------------------------
__global__ __launch_bounds__(256)
void wmma_gemm_bias_k(const _Float16* __restrict__ A,   // M x K row-major
                      const _Float16* __restrict__ Bt,  // N x K (K contiguous)
                      const float* __restrict__ bias,   // N or nullptr
                      float* __restrict__ C,            // M x N
                      int M, int K, int N) {
  __shared__ _Float16 As[128][40];  // pad 32->40 halves (80B rows, 16B aligned)
  __shared__ _Float16 Bs[64][40];
  const int tid  = threadIdx.x;
  const int wave = tid >> 5;
  const int lane = tid & 31;
  const int m0 = blockIdx.x * 128;
  const int n0 = blockIdx.y * 64;
  const int wm = wave & 3;        // 0..3 -> 32-row band
  const int wn = wave >> 2;       // 0..1 -> 32-col band

  const bool mInt = (m0 + 128) <= M;
  const bool nInt = (n0 + 64) <= N;

  v8f acc00 = {}, acc01 = {}, acc10 = {}, acc11 = {};

  for (int k0 = 0; k0 < K; k0 += 32) {
    const bool kFull = (k0 + 32) <= K;

    // ---- stage A tile 128x32 ----
    if (mInt && kFull) {
#pragma unroll
      for (int j = 0; j < 2; ++j) {
        int cidx = tid + 256 * j;            // 512 chunks of 8 halves
        int r  = cidx >> 2;                  // 0..127
        int c8 = (cidx & 3) * 8;
        const _Float16* src = A + (long long)(m0 + r) * K + k0 + c8;
#if USE_ASYNC_LDS
        ASYNC_COPY_B128(&As[r][c8], src);
#else
        *(float4*)&As[r][c8] = *(const float4*)src;
#endif
      }
    } else {
      for (int e = tid; e < 128 * 32; e += 256) {
        int r = e >> 5, kk = e & 31;
        int gm = m0 + r, gk = k0 + kk;
        As[r][kk] = (gm < M && gk < K) ? A[(long long)gm * K + gk] : (_Float16)0.f;
      }
    }
    // ---- stage B tile: 64 columns x 32 K-values ----
    if (nInt && kFull) {
      int r  = tid >> 2;                     // 0..63
      int c8 = (tid & 3) * 8;
      const _Float16* src = Bt + (long long)(n0 + r) * K + k0 + c8;
#if USE_ASYNC_LDS
      ASYNC_COPY_B128(&Bs[r][c8], src);
#else
      *(float4*)&Bs[r][c8] = *(const float4*)src;
#endif
    } else {
      for (int e = tid; e < 64 * 32; e += 256) {
        int c = e >> 5, kk = e & 31;
        int gn = n0 + c, gk = k0 + kk;
        Bs[c][kk] = (gn < N && gk < K) ? Bt[(long long)gn * K + gk] : (_Float16)0.f;
      }
    }
#if USE_ASYNC_LDS
    __builtin_amdgcn_s_wait_asynccnt(0);
#endif
    __syncthreads();

    // A fragments (ISA 7.12.2, 16-bit A 16x32): lanes 0-15 rows 0-15 hold
    // K {0..7,16..23}; lanes 16-31 hold K {8..15,24..31} of the same rows.
    const int hs = (lane >> 4) * 8;
    const int ar0 = wm * 32 + (lane & 15);
    const int ar1 = ar0 + 16;
    v16h a0, a1;
#pragma unroll
    for (int i = 0; i < 8; ++i) {
      a0[i]     = As[ar0][hs + i];
      a0[8 + i] = As[ar0][16 + hs + i];
      a1[i]     = As[ar1][hs + i];
      a1[8 + i] = As[ar1][16 + hs + i];
    }
    // B fragments (32x16): lanes 0-15 hold K {0..7,16..23} of col n, lanes 16-31 K {8..15,24..31}.
    const int ks = (lane >> 4) * 8;
    const int bc0 = wn * 32 + (lane & 15);
    const int bc1 = bc0 + 16;
    v16h b0, b1;
#pragma unroll
    for (int i = 0; i < 8; ++i) {
      b0[i]     = Bs[bc0][ks + i];
      b0[8 + i] = Bs[bc0][16 + ks + i];
      b1[i]     = Bs[bc1][ks + i];
      b1[8 + i] = Bs[bc1][16 + ks + i];
    }

    acc00 = __builtin_amdgcn_wmma_f32_16x16x32_f16(false, a0, false, b0, (short)0, acc00, false, false);
    acc01 = __builtin_amdgcn_wmma_f32_16x16x32_f16(false, a0, false, b1, (short)0, acc01, false, false);
    acc10 = __builtin_amdgcn_wmma_f32_16x16x32_f16(false, a1, false, b0, (short)0, acc10, false, false);
    acc11 = __builtin_amdgcn_wmma_f32_16x16x32_f16(false, a1, false, b1, (short)0, acc11, false, false);
    __syncthreads();
  }

  // C/D layout: VGPR r -> row (r + 8*(lane/16)), col = lane%16
  const int rb0 = m0 + wm * 32 + ((lane >> 4) << 3);
  const int rb1 = rb0 + 16;
  const int c0 = n0 + wn * 32 + (lane & 15);
  const int c1 = c0 + 16;
  const float bv0 = (bias && c0 < N) ? bias[c0] : 0.f;
  const float bv1 = (bias && c1 < N) ? bias[c1] : 0.f;
#pragma unroll
  for (int r = 0; r < 8; ++r) {
    int g0 = rb0 + r, g1 = rb1 + r;
    if (g0 < M) {
      if (c0 < N) C[(long long)g0 * N + c0] = acc00[r] + bv0;
      if (c1 < N) C[(long long)g0 * N + c1] = acc01[r] + bv1;
    }
    if (g1 < M) {
      if (c0 < N) C[(long long)g1 * N + c0] = acc10[r] + bv0;
      if (c1 < N) C[(long long)g1 * N + c1] = acc11[r] + bv1;
    }
  }
}

// ---------------------------------------------------------------------------
// BatchNorm: per-column mean / rsqrt(var+eps)  (train-mode batch stats)
// ---------------------------------------------------------------------------
__global__ void colstats_k(const float* __restrict__ X, int M, int N,
                           float* __restrict__ mean, float* __restrict__ rstd) {
  const int col = blockIdx.x;
  __shared__ float s1[256], s2[256];
  float a = 0.f, b = 0.f;
  for (int r = threadIdx.x; r < M; r += 256) {
    float v = X[(long long)r * N + col];
    a += v; b += v * v;
  }
  s1[threadIdx.x] = a; s2[threadIdx.x] = b;
  __syncthreads();
  for (int s = 128; s > 0; s >>= 1) {
    if (threadIdx.x < s) { s1[threadIdx.x] += s1[threadIdx.x + s]; s2[threadIdx.x] += s2[threadIdx.x + s]; }
    __syncthreads();
  }
  if (threadIdx.x == 0) {
    float m = s1[0] / (float)M;
    float v = s2[0] / (float)M - m * m;
    mean[col] = m;
    rstd[col] = rsqrtf(v + BN_EPS);
  }
}

__global__ void bn_apply_k(float* __restrict__ X, long long total, int N,
                           const float* __restrict__ mean, const float* __restrict__ rstd,
                           const float* __restrict__ gamma, const float* __restrict__ beta,
                           int act) {
  long long i = (long long)blockIdx.x * blockDim.x + threadIdx.x;
  if (i >= total) return;
  int c = (int)(i % N);
  float v = (X[i] - mean[c]) * rstd[c] * gamma[c] + beta[c];
  X[i] = act ? leaky(v) : v;
}

__global__ void add_leaky_k(const float* __restrict__ a, const float* __restrict__ b,
                            float* __restrict__ o, long long n) {
  long long i = (long long)blockIdx.x * blockDim.x + threadIdx.x;
  if (i < n) o[i] = leaky(a[i] + b[i]);
}

// ---------------------------------------------------------------------------
// kNN (k=16, self included) — register insertion sort over all points
// ---------------------------------------------------------------------------
__global__ void knn16_k(const float* __restrict__ pos, int n, int* __restrict__ idx) {
  int q = blockIdx.x * blockDim.x + threadIdx.x;
  if (q >= n) return;
  const float qx = pos[q * 3], qy = pos[q * 3 + 1], qz = pos[q * 3 + 2];
  float bd[KNN_K]; int bi[KNN_K];
#pragma unroll
  for (int i = 0; i < KNN_K; ++i) { bd[i] = 3.4e38f; bi[i] = 0; }
  for (int p = 0; p < n; ++p) {
    __builtin_prefetch(&pos[(p + 64) * 3], 0, 0);
    float dx = pos[p * 3] - qx, dy = pos[p * 3 + 1] - qy, dz = pos[p * 3 + 2] - qz;
    float d = dx * dx + dy * dy + dz * dz;
    if (d < bd[KNN_K - 1]) {
      bd[KNN_K - 1] = d; bi[KNN_K - 1] = p;
#pragma unroll
      for (int t = KNN_K - 1; t > 0; --t) {
        if (bd[t] < bd[t - 1]) {
          float td = bd[t]; bd[t] = bd[t - 1]; bd[t - 1] = td;
          int   ti = bi[t]; bi[t] = bi[t - 1]; bi[t - 1] = ti;
        }
      }
    }
  }
#pragma unroll
  for (int i = 0; i < KNN_K; ++i) idx[q * KNN_K + i] = bi[i];
}

// nearest source point per query (knn_interpolate k=1)
__global__ void nn1_k(const float* __restrict__ src, int ns,
                      const float* __restrict__ q, int nq, int* __restrict__ out) {
  int i = blockIdx.x * blockDim.x + threadIdx.x;
  if (i >= nq) return;
  const float qx = q[i * 3], qy = q[i * 3 + 1], qz = q[i * 3 + 2];
  float best = 3.4e38f; int bi = 0;
  for (int p = 0; p < ns; ++p) {
    float dx = src[p * 3] - qx, dy = src[p * 3 + 1] - qy, dz = src[p * 3 + 2] - qz;
    float d = dx * dx + dy * dy + dz * dz;
    if (d < best) { best = d; bi = p; }
  }
  out[i] = bi;
}

// ---------------------------------------------------------------------------
// LFA helpers
// ---------------------------------------------------------------------------
__global__ void build_rel_k(const float* __restrict__ pos, const int* __restrict__ idx,
                            int n, float* __restrict__ rel) {
  int i = blockIdx.x * blockDim.x + threadIdx.x;
  if (i >= n * KNN_K) return;
  int pt = i >> 4;
  int j = idx[i];
  float ix = pos[pt * 3], iy = pos[pt * 3 + 1], iz = pos[pt * 3 + 2];
  float jx = pos[j * 3],  jy = pos[j * 3 + 1],  jz = pos[j * 3 + 2];
  float dx = jx - ix, dy = jy - iy, dz = jz - iz;
  float dist = sqrtf(dx * dx + dy * dy + dz * dz);
  float* o = rel + (long long)i * 10;
  o[0] = ix; o[1] = iy; o[2] = iz;
  o[3] = jx; o[4] = jy; o[5] = jz;
  o[6] = dx; o[7] = dy; o[8] = dz; o[9] = dist;
}

// lf[:, :h] = x[idx]; lf[:, h:] = lse   (rows = n*K, width c = 2h)
__global__ void gather_concat_k(const float* __restrict__ x, const int* __restrict__ idx,
                                const float* __restrict__ lse, long long rows, int h,
                                float* __restrict__ lf) {
  int c = 2 * h;
  long long i = (long long)blockIdx.x * blockDim.x + threadIdx.x;
  if (i >= rows * c) return;
  long long r = i / c; int ch = (int)(i % c);
  lf[i] = (ch < h) ? x[(long long)idx[r] * h + ch] : lse[r * h + (ch - h)];
}

// out[n,ch] = sum_k softmax_k(att[n,k,ch]) * lf[n,k,ch]
__global__ void attn_agg_k(const float* __restrict__ lf, const float* __restrict__ att,
                           int n, int c, float* __restrict__ out) {
  long long i = (long long)blockIdx.x * blockDim.x + threadIdx.x;
  if (i >= (long long)n * c) return;
  long long pt = i / c; int ch = (int)(i % c);
  const float* ar = att + pt * KNN_K * c + ch;
  const float* lr = lf  + pt * KNN_K * c + ch;
  float mx = -3.4e38f;
#pragma unroll
  for (int k = 0; k < KNN_K; ++k) mx = fmaxf(mx, ar[k * c]);
  float s = 0.f, acc = 0.f;
#pragma unroll
  for (int k = 0; k < KNN_K; ++k) {
    float e = __expf(ar[k * c] - mx);
    s += e;
    acc += e * lr[k * c];
  }
  out[i] = acc / s;
}

// ---------------------------------------------------------------------------
// decimation / gathers / concat / output
// ---------------------------------------------------------------------------
__global__ void perm_idx_k(int n, int m, unsigned mult, unsigned add, int* __restrict__ out) {
  int j = blockIdx.x * blockDim.x + threadIdx.x;
  if (j < m) out[j] = (int)(((unsigned)j * mult + add) & (unsigned)(n - 1));
}

__global__ void gather_rows_k(const float* __restrict__ src, const int* __restrict__ idx,
                              float* __restrict__ dst, long long rows, int c) {
  long long i = (long long)blockIdx.x * blockDim.x + threadIdx.x;
  if (i >= rows * c) return;
  long long r = i / c; int ch = (int)(i % c);
  dst[i] = src[(long long)idx[r] * c + ch];
}

__global__ void concat2_k(const float* __restrict__ a, const float* __restrict__ b,
                          float* __restrict__ o, long long rows, int c1, int c2) {
  int c = c1 + c2;
  long long i = (long long)blockIdx.x * blockDim.x + threadIdx.x;
  if (i >= rows * c) return;
  long long r = i / c; int ch = (int)(i % c);
  o[i] = (ch < c1) ? a[r * c1 + ch] : b[r * c2 + (ch - c1)];
}

__global__ void logsoftmax_k(const float* __restrict__ L, int n, int c, float* __restrict__ out) {
  int i = blockIdx.x * blockDim.x + threadIdx.x;
  if (i >= n) return;
  const float* r = L + (long long)i * c;
  float mx = -3.4e38f;
  for (int j = 0; j < c; ++j) mx = fmaxf(mx, r[j]);
  float s = 0.f;
  for (int j = 0; j < c; ++j) s += __expf(r[j] - mx);
  float ls = __logf(s);
  float* o = out + (long long)i * c;
  for (int j = 0; j < c; ++j) o[j] = r[j] - mx - ls;
}

// ---------------------------------------------------------------------------
// Host orchestration
// ---------------------------------------------------------------------------
struct MlpLayer { const float* W; const float* b; const float* beta; const float* gamma; int in, out; };
struct LfaP     { const float* attW; MlpLayer enc; MlpLayer post; int c; };
struct BlockP   { LfaP lfa1, lfa2; MlpLayer mlp1, mlp2, shortcut; };

static inline long long cdivll(long long a, long long b) { return (a + b - 1) / b; }

extern "C" void kernel_launch(void* const* d_in, const int* in_sizes, int n_in,
                              void* d_out, int out_size, void* d_ws, size_t ws_size,
                              hipStream_t stream) {
  (void)in_sizes; (void)n_in; (void)out_size; (void)ws_size;

  const int N1 = 32768, N2 = 8192, N3 = 2048, N4 = 512;
  const int N1d = N1 / 4, N2d = N2 / 4, N3d = N3 / 4, N4d = N4 / 4; // 8192,2048,512,128

  const float* x_in = (const float*)d_in[0];
  const float* pos  = (const float*)d_in[1];
  // d_in[2..4]: batch / ptr / cluster_id (unused, single cluster)

  // ---- params: flattened pytree, dict keys sorted alphabetically ----------
  int cur = 5;
  auto nextp = [&]() { return (const float*)d_in[cur++]; };
  auto mlayer = [&](int i, int o) {
    MlpLayer L; L.W = nextp(); L.b = nextp(); L.beta = nextp(); L.gamma = nextp();
    L.in = i; L.out = o; return L;
  };
  auto lfaP = [&](int c) {                 // keys: attW, enc, post
    LfaP f; f.attW = nextp(); f.enc = mlayer(10, c / 2); f.post = mlayer(c, c); f.c = c; return f;
  };
  auto blockP = [&](int di, int dout) {    // keys: lfa1, lfa2, mlp1, mlp2, shortcut
    BlockP b;
    b.lfa1 = lfaP(dout / 4); b.lfa2 = lfaP(dout / 2);
    b.mlp1 = mlayer(di, dout / 8); b.mlp2 = mlayer(dout / 2, dout);
    b.shortcut = mlayer(di, dout);
    return b;
  };
  // top-level sorted: b1,b2,b3,b4,classif,fc0,fc_out,fp1,fp2,fp3,fp4,summit
  BlockP B1 = blockP(32, 32);
  BlockP B2 = blockP(32, 128);
  BlockP B3 = blockP(128, 256);
  BlockP B4 = blockP(256, 512);
  MlpLayer CL0 = mlayer(32, 64), CL1 = mlayer(64, 32);
  const float* fc0W = nextp();  const float* fc0b = nextp();
  const float* fcoW = nextp();  const float* fcob = nextp();
  MlpLayer FP1 = mlayer(64, 32);
  MlpLayer FP2 = mlayer(160, 32);
  MlpLayer FP3 = mlayer(384, 128);
  MlpLayer FP4 = mlayer(768, 256);
  MlpLayer SUM = mlayer(512, 512);

  // ---- workspace bump allocator ------------------------------------------
  char* wp = (char*)d_ws;
  auto alloc = [&](size_t bytes) {
    void* p = wp;
    wp += (bytes + 255) & ~(size_t)255;
    return p;
  };
  auto af = [&](size_t n) { return (float*)alloc(n * sizeof(float)); };
  auto ah = [&](size_t n) { return (_Float16*)alloc(n * sizeof(_Float16)); };
  auto ai = [&](size_t n) { return (int*)alloc(n * sizeof(int)); };

  // persistent activations
  float* X0  = af((size_t)N1 * 32);
  float* X1  = af((size_t)N1 * 32);
  float* X1D = af((size_t)N1d * 32);  float* P1D = af((size_t)N1d * 3);
  float* X2  = af((size_t)N2 * 128);
  float* X2D = af((size_t)N2d * 128); float* P2D = af((size_t)N2d * 3);
  float* X3  = af((size_t)N3 * 256);
  float* X3D = af((size_t)N3d * 256); float* P3D = af((size_t)N3d * 3);
  float* X4  = af((size_t)N4 * 512);
  float* X4D = af((size_t)N4d * 512); float* P4D = af((size_t)N4d * 3);
  float* XS  = af((size_t)N4d * 512);
  float* F4  = af((size_t)N3d * 256);
  float* F3  = af((size_t)N2d * 128);
  float* F2  = af((size_t)N1d * 32);
  float* F1  = af((size_t)N1 * 32);
  float* H1  = af((size_t)N1 * 64);
  float* H2  = af((size_t)N1 * 32);
  float* LOG = af((size_t)N1 * 7);

  // reusable scratch (sized to maxima across the network)
  const size_t MAXLF = (size_t)N1 * KNN_K * 16;          // 8.39M (== b2 lfa2)
  float* REL = af((size_t)N1 * KNN_K * 10);
  float* LSE = af(MAXLF / 2);
  float* LF  = af(MAXLF);
  float* ATT = af(MAXLF);
  float* SC  = af((size_t)N1 * 32);                      // max n*dout = 1.05M
  float* T1  = af((size_t)N1 * 32);
  float* T2  = af((size_t)N1 * 32);
  float* T3  = af((size_t)N1 * 32);
  float* T4  = af((size_t)N1 * 32);
  float* AGG = af((size_t)N1 * 32);
  float* XI  = af((size_t)2 * 1048576);
  float* CC  = af((size_t)2 * 2097152);
  float* MEAN = af(1024);
  float* RSTD = af(1024);
  _Float16* hA = ah(MAXLF);                              // activation f16 arena
  _Float16* hW = ah(262144);                             // weight f16 arena
  int* IDX  = ai((size_t)N1 * KNN_K);
  int* PIDX = ai(N1d);
  int* NNI  = ai(N1);

  // ---- launch helpers -----------------------------------------------------
  auto gemm = [&](const float* X, const float* W, const float* bias, float* Y,
                  long long M, int K, int N) {
    long long nx = M * K;
    f32_to_f16_k<<<(unsigned)cdivll(nx, 256), 256, 0, stream>>>(X, hA, nx);
    w_to_f16t_k<<<(unsigned)cdivll((long long)K * N, 256), 256, 0, stream>>>(W, hW, K, N);
    dim3 g((unsigned)cdivll(M, 128), (unsigned)cdivll(N, 64));
    wmma_gemm_bias_k<<<g, 256, 0, stream>>>(hA, hW, bias, Y, (int)M, K, N);
  };
  auto smlp1 = [&](const float* X, long long M, const MlpLayer& L, float* Y, bool act) {
    gemm(X, L.W, L.b, Y, M, L.in, L.out);
    colstats_k<<<L.out, 256, 0, stream>>>(Y, (int)M, L.out, MEAN, RSTD);
    long long t = M * L.out;
    bn_apply_k<<<(unsigned)cdivll(t, 256), 256, 0, stream>>>(Y, t, L.out, MEAN, RSTD,
                                                             L.gamma, L.beta, act ? 1 : 0);
  };
  auto run_lfa = [&](const LfaP& f, const int* idx, const float* x, const float* p,
                     int n, float* out) {
    int c = f.c, h = c / 2;
    build_rel_k<<<(unsigned)cdivll((long long)n * KNN_K, 256), 256, 0, stream>>>(p, idx, n, REL);
    smlp1(REL, (long long)n * KNN_K, f.enc, LSE, true);
    long long rows = (long long)n * KNN_K;
    gather_concat_k<<<(unsigned)cdivll(rows * c, 256), 256, 0, stream>>>(x, idx, LSE, rows, h, LF);
    gemm(LF, f.attW, nullptr, ATT, rows, c, c);          // mlp_attention: no bias/norm/act
    attn_agg_k<<<(unsigned)cdivll((long long)n * c, 256), 256, 0, stream>>>(LF, ATT, n, c, AGG);
    smlp1(AGG, n, f.post, out, true);
  };
  auto run_block = [&](const BlockP& B, const float* x, const float* p, int n, int dout,
                       float* out) {
    knn16_k<<<(unsigned)cdivll(n, 256), 256, 0, stream>>>(p, n, IDX);
    smlp1(x, n, B.shortcut, SC, false);
    smlp1(x, n, B.mlp1, T1, true);
    run_lfa(B.lfa1, IDX, T1, p, n, T2);
    run_lfa(B.lfa2, IDX, T2, p, n, T3);
    smlp1(T3, n, B.mlp2, T4, false);
    add_leaky_k<<<(unsigned)cdivll((long long)n * dout, 256), 256, 0, stream>>>(
        T4, SC, out, (long long)n * dout);
  };
  auto decimate = [&](const float* x, const float* p, int n, int c,
                      float* xd, float* pd, unsigned salt) {
    int m = n / 4;
    perm_idx_k<<<(unsigned)cdivll(m, 256), 256, 0, stream>>>(n, m, 2654435761u, salt, PIDX);
    gather_rows_k<<<(unsigned)cdivll((long long)m * c, 256), 256, 0, stream>>>(x, PIDX, xd, m, c);
    gather_rows_k<<<(unsigned)cdivll((long long)m * 3, 256), 256, 0, stream>>>(p, PIDX, pd, m, 3);
  };
  auto run_fp = [&](const MlpLayer& L, const float* xlow, const float* plow, int nlow,
                    const float* xskip, const float* pskip, int nskip,
                    int clow, int cskip, float* out) {
    nn1_k<<<(unsigned)cdivll(nskip, 256), 256, 0, stream>>>(plow, nlow, pskip, nskip, NNI);
    gather_rows_k<<<(unsigned)cdivll((long long)nskip * clow, 256), 256, 0, stream>>>(
        xlow, NNI, XI, nskip, clow);
    concat2_k<<<(unsigned)cdivll((long long)nskip * (clow + cskip), 256), 256, 0, stream>>>(
        XI, xskip, CC, nskip, clow, cskip);
    smlp1(CC, nskip, L, out, true);
  };

  // ---- forward ------------------------------------------------------------
  gemm(x_in, fc0W, fc0b, X0, N1, 9, 32);                 // fc0 (no BN/act)

  run_block(B1, X0, pos, N1, 32, X1);
  decimate(X1, pos, N1, 32, X1D, P1D, 0x9E3779B1u);

  run_block(B2, X1D, P1D, N2, 128, X2);
  decimate(X2, P1D, N2, 128, X2D, P2D, 0x85EBCA77u);

  run_block(B3, X2D, P2D, N3, 256, X3);
  decimate(X3, P2D, N3, 256, X3D, P3D, 0xC2B2AE3Du);

  run_block(B4, X3D, P3D, N4, 512, X4);
  decimate(X4, P3D, N4, 512, X4D, P4D, 0x27D4EB2Fu);

  smlp1(X4D, N4d, SUM, XS, true);                        // summit

  run_fp(FP4, XS, P4D, N4d, X3D, P3D, N3d, 512, 256, F4);
  run_fp(FP3, F4, P3D, N3d, X2D, P2D, N2d, 256, 128, F3);
  run_fp(FP2, F3, P2D, N2d, X1D, P1D, N1d, 128, 32, F2);
  run_fp(FP1, F2, P1D, N1d, X1, pos, N1, 32, 32, F1);

  smlp1(F1, N1, CL0, H1, true);                          // classif
  smlp1(H1, N1, CL1, H2, true);
  gemm(H2, fcoW, fcob, LOG, N1, 32, 7);                  // fc_out (no BN/act)

  logsoftmax_k<<<(unsigned)cdivll(N1, 256), 256, 0, stream>>>(LOG, N1, 7, (float*)d_out);
}
